// SpaceSelfAttention_33423435497980
// MI455X (gfx1250) — matrix operand order repaired and checked by reading, not compiled, not run on hardware
//
#include <hip/hip_runtime.h>
#include <hip/hip_bf16.h>

// ---------------------------------------------------------------------------
// Types for CDNA5 WMMA (wave32): v_wmma_f32_16x16x32_bf16
// ---------------------------------------------------------------------------
typedef __bf16 bf16;
typedef __attribute__((ext_vector_type(16))) bf16  v16bf;
typedef __attribute__((ext_vector_type(8)))  bf16  v8bf;
typedef __attribute__((ext_vector_type(4)))  bf16  v4bf;
typedef __attribute__((ext_vector_type(8)))  float v8f;

#define BATCH   16
#define TSEQ    1033
#define TPAD    1088        // 17*64, multiple of 32-key tiles; covers all q blocks
#define CDIM    768
#define NHEAD   12
#define HDIM    64
#define MROWS   (BATCH * TSEQ)   // 16528 (= 1033 * 16, divisible by 16)
#define ROPE_OFFSET 9

static __device__ __forceinline__ v8f wmma_bf16(v16bf a, v16bf b, v8f c) {
    return __builtin_amdgcn_wmma_f32_16x16x32_bf16(
        /*neg_a=*/false, a, /*neg_b=*/false, b,
        /*c_mod=*/(short)0, c, /*reuse_a=*/false, /*reuse_b=*/false);
}

static __device__ __forceinline__ v16bf cat8(v8bf lo, v8bf hi) {
    return __builtin_shufflevector(lo, hi, 0,1,2,3,4,5,6,7,8,9,10,11,12,13,14,15);
}

// ---------------------------------------------------------------------------
// f32 -> bf16 elementwise convert (n divisible by 4)
// ---------------------------------------------------------------------------
__global__ void __launch_bounds__(256)
k_f32_to_bf16(const float* __restrict__ src, bf16* __restrict__ dst, int n4) {
    int i = blockIdx.x * 256 + threadIdx.x;
    if (i >= n4) return;
    const float4 v = reinterpret_cast<const float4*>(src)[i];
    v4bf o = { (bf16)v.x, (bf16)v.y, (bf16)v.z, (bf16)v.w };
    reinterpret_cast<v4bf*>(dst)[i] = o;
}

// zero fill bf16 (n divisible by 8)
__global__ void __launch_bounds__(256)
k_zero_bf16(bf16* __restrict__ dst, int n8) {
    int i = blockIdx.x * 256 + threadIdx.x;
    if (i >= n8) return;
    v8bf z = {};
    reinterpret_cast<v8bf*>(dst)[i] = z;
}

// ---------------------------------------------------------------------------
// GEMM: C[M,N](f32) = A[M,K](bf16, row-major) * W[N,K](bf16, row-major)^T
// Block = 4 waves (128 thr). Wave computes a 16x64 strip: 4 accumulators.
// The shared 16x64 A-tile (2 KB) is staged via CDNA5 async global->LDS DMA
// (global_load_async_to_lds_b128, ASYNCcnt), double-buffered: the fetch of
// tile i+1 overlaps the 8 WMMAs of tile i. Next-tile issue is unconditional
// (last iteration redundantly refetches into the dead buffer) and W prefetch
// is unconditional speculative, so the hot loop is branch-free.
// grid = (M/16, N/256)
// ---------------------------------------------------------------------------
__global__ void __launch_bounds__(128)
k_gemm_bf16nt(const bf16* __restrict__ A, const bf16* __restrict__ W,
              float* __restrict__ C, int M, int N, int K) {
    __shared__ bf16 atile[2][16][64];   // double-buffered A tile (4 KB)

    const int tid  = threadIdx.x;
    const int lane = tid & 31;
    const int wave = tid >> 5;
    const int half = lane >> 4;
    const int l16  = lane & 15;
    const int m0   = blockIdx.x * 16;
    const int n0   = (blockIdx.y * 4 + wave) * 64;

    // cooperative async A fetch: 128 lanes x 16B = full 16x128B tile
    const int arow = tid >> 3;              // 0..15
    const int acol = (tid & 7) * 8;         // element offset (8 bf16 = 16 B)
    const unsigned abase = (unsigned)(((unsigned)(m0 + arow) * (unsigned)K + acol) * 2u);
    unsigned lds_dst[2];
    lds_dst[0] = (unsigned)(uintptr_t)&atile[0][arow][acol];
    lds_dst[1] = (unsigned)(uintptr_t)&atile[1][arow][acol];

    // prologue: fetch tile 0
    asm volatile("global_load_async_to_lds_b128 %0, %1, %2"
                 :: "v"(lds_dst[0]), "v"(abase), "s"(A) : "memory");

    v8f acc[4] = {};
    const int nsteps = K / 64;
    for (int i = 0; i < nsteps; ++i) {
        const int kb = i * 64;
        asm volatile("s_wait_asynccnt 0x0" ::: "memory");  // my slice of tile i landed
        __syncthreads();                                   // all slices visible; all
                                                           // iter i-1 LDS reads done
        // overlap: kick off tile i+1 (clamped; dead refetch on last iter)
        {
            const int nkb = (kb + 64 < K) ? (kb + 64) : kb;
            unsigned voff = abase + (unsigned)(nkb * 2);
            asm volatile("global_load_async_to_lds_b128 %0, %1, %2"
                         :: "v"(lds_dst[(i + 1) & 1]), "v"(voff), "s"(A) : "memory");
        }
#pragma unroll
        for (int ks = 0; ks < 2; ++ks) {
            // A fragment from LDS in WMMA A-layout
            v8bf alo = *(const v8bf*)&atile[i & 1][l16][ks * 32 + half * 16];
            v8bf ahi = *(const v8bf*)&atile[i & 1][l16][ks * 32 + half * 16 + 8];
            v16bf a = cat8(alo, ahi);
#pragma unroll
            for (int g = 0; g < 4; ++g) {
                const bf16* bp = W + (size_t)(n0 + g * 16 + l16) * K
                                   + kb + ks * 32 + half * 16;
                if (ks == 0)                               // compile-time, no branch
                    __builtin_prefetch(bp + 512, 0, 0);    // speculative, ~1KB ahead
                v16bf b = *(const v16bf*)bp;
                acc[g] = wmma_bf16(a, b, acc[g]);
            }
        }
    }
#pragma unroll
    for (int g = 0; g < 4; ++g)
#pragma unroll
        for (int j = 0; j < 8; ++j) {
            const int row = j + half * 8;  // C/D layout: VGPR j -> M=j (lanes<16) / j+8
            C[(size_t)(m0 + row) * N + n0 + g * 16 + l16] = acc[g][j];
        }
}

// ---------------------------------------------------------------------------
// Fused LayerNorm + axial RoPE + bf16 pack into head-major [B,H,TPAD,64].
// Block = 192 threads, one (b,t) row; each thread owns 4 contiguous channels
// (= 2 rotation pairs, never split across threads).
// ---------------------------------------------------------------------------
__global__ void __launch_bounds__(192)
k_ln_rope(const float* __restrict__ src, const float* __restrict__ w,
          const float* __restrict__ bia, const float* __restrict__ fcos,
          const float* __restrict__ fsin, bf16* __restrict__ dst,
          float outscale) {
    const int bt  = blockIdx.x;
    const int tid = threadIdx.x;
    const int c0  = tid * 4;

    const float4 xv = *(const float4*)(src + (size_t)bt * CDIM + c0);
    float s  = xv.x + xv.y + xv.z + xv.w;
    float ss = xv.x * xv.x + xv.y * xv.y + xv.z * xv.z + xv.w * xv.w;
#pragma unroll
    for (int m = 1; m < 32; m <<= 1) {
        s  += __shfl_xor(s,  m, 32);
        ss += __shfl_xor(ss, m, 32);
    }
    __shared__ float2 wred[6];
    const int lane = tid & 31, wid = tid >> 5;
    if (lane == 0) wred[wid] = make_float2(s, ss);
    __syncthreads();
    float ts = 0.f, tss = 0.f;
#pragma unroll
    for (int i = 0; i < 6; ++i) { ts += wred[i].x; tss += wred[i].y; }
    const float inv = 1.0f / (float)CDIM;
    const float mean = ts * inv;
    const float var  = tss * inv - mean * mean;
    const float rstd = rsqrtf(var + 1e-5f);

    const float4 wv = *(const float4*)(w + c0);
    const float4 bv = *(const float4*)(bia + c0);
    float n0 = (xv.x - mean) * rstd * wv.x + bv.x;
    float n1 = (xv.y - mean) * rstd * wv.y + bv.y;
    float n2 = (xv.z - mean) * rstd * wv.z + bv.z;
    float n3 = (xv.w - mean) * rstd * wv.w + bv.w;

    const int t = bt % TSEQ;
    const int b = bt / TSEQ;
    float o0 = n0, o1 = n1, o2 = n2, o3 = n3;
    if (t >= ROPE_OFFSET) {                 // uniform branch per block
        const int p  = t - ROPE_OFFSET;     // < 1024
        const int i0 = (c0 & 63) >> 1;      // freq index for first pair (even)
        const float c_0 = fcos[p * 32 + i0],     s_0 = fsin[p * 32 + i0];
        const float c_1 = fcos[p * 32 + i0 + 1], s_1 = fsin[p * 32 + i0 + 1];
        o0 = n0 * c_0 - n1 * s_0;  o1 = n0 * s_0 + n1 * c_0;
        o2 = n2 * c_1 - n3 * s_1;  o3 = n2 * s_1 + n3 * c_1;
    }
    const int h = c0 >> 6;
    bf16* d = dst + (((size_t)(b * NHEAD + h) * TPAD) + t) * HDIM + (c0 & 63);
    v4bf ov = { (bf16)(o0 * outscale), (bf16)(o1 * outscale),
                (bf16)(o2 * outscale), (bf16)(o3 * outscale) };
    *(v4bf*)d = ov;
}

// ---------------------------------------------------------------------------
// V transpose: v_raw[B*T, C] f32 -> vT[B,H,64,TPAD] bf16 (pad pre-zeroed)
// ---------------------------------------------------------------------------
__global__ void __launch_bounds__(256)
k_transpose_v(const float* __restrict__ vr, bf16* __restrict__ vt, long total) {
    long gid = (long)blockIdx.x * 256 + threadIdx.x;  // over B*H*64*TSEQ
    if (gid >= total) return;
    const int  t  = (int)(gid % TSEQ);
    const long r  = gid / TSEQ;              // (b*H + h)*64 + d
    const int  d  = (int)(r % HDIM);
    const int  bh = (int)(r / HDIM);
    const int  b  = bh / NHEAD, h = bh % NHEAD;
    const float val = vr[((size_t)b * TSEQ + t) * CDIM + h * HDIM + d];
    vt[(size_t)r * TPAD + t] = (bf16)val;
}

// ---------------------------------------------------------------------------
// Flash-style attention. Block = 4 waves; each wave owns a 16-row Q tile of
// one (b,h); streams 32-key tiles: 4 WMMAs for S = Q*K^T, online softmax via
// 16-lane shuffles, P converted C-layout -> A-layout through a *wave-private*
// LDS tile. Intra-wave LDS ops complete in order (DScnt), so a wave-local
// s_wait_dscnt fence replaces block barriers entirely.
// grid = (TPAD/64, B*H)
// ---------------------------------------------------------------------------
__global__ void __launch_bounds__(128)
k_attention(const bf16* __restrict__ Q, const bf16* __restrict__ Kt,
            const bf16* __restrict__ Vt, bf16* __restrict__ Y) {
    __shared__ bf16 ptile[4][16][32];

    const int lane = threadIdx.x & 31;
    const int wave = threadIdx.x >> 5;
    const int half = lane >> 4;
    const int l16  = lane & 15;
    const int bh   = blockIdx.y;
    const int b    = bh / NHEAD, h = bh % NHEAD;
    const int m0   = blockIdx.x * 64 + wave * 16;   // q-tile base row (padded)

    // Q tile in A-layout: two K-chunks covering d = 0..31 and 32..63
    const bf16* qrow = Q + ((size_t)bh * TPAD + m0 + l16) * HDIM;
    v16bf qa[2];
#pragma unroll
    for (int c = 0; c < 2; ++c) {
        v8bf lo = *(const v8bf*)(qrow + c * 32 + half * 16);
        v8bf hi = *(const v8bf*)(qrow + c * 32 + half * 16 + 8);
        qa[c] = cat8(lo, hi);
    }

    v8f O[4] = {};
    float mr[8], lr[8];
#pragma unroll
    for (int j = 0; j < 8; ++j) { mr[j] = -1e30f; lr[j] = 0.f; }

    const bf16* kbase = Kt + (size_t)bh * TPAD * HDIM;
    const bf16* vbase = Vt + (size_t)bh * HDIM * TPAD;

    for (int kt = 0; kt < TPAD; kt += 32) {
        // --- scores: two 16x16 n-tiles, K=64 in two bf16 WMMAs each ---
        v8f s[2] = {};
#pragma unroll
        for (int nt = 0; nt < 2; ++nt) {
            const bf16* kr = kbase + (size_t)(kt + nt * 16 + l16) * HDIM + half * 16;
            v16bf b0 = *(const v16bf*)(kr);        // d 0..31
            v16bf b1 = *(const v16bf*)(kr + 32);   // d 32..63
            s[nt] = wmma_bf16(qa[0], b0, s[nt]);
            s[nt] = wmma_bf16(qa[1], b1, s[nt]);
            const int key = kt + nt * 16 + l16;
            const float pen = (key >= TSEQ) ? -1e30f : 0.0f;  // mask pad keys
            s[nt] = s[nt] + pen;                                // lane-uniform over j
        }
        // --- online softmax; C-layout rows align with lane groups ---
#pragma unroll
        for (int j = 0; j < 8; ++j) {
            float t = fmaxf(s[0][j], s[1][j]);
            t = fmaxf(t, __shfl_xor(t, 1, 32));
            t = fmaxf(t, __shfl_xor(t, 2, 32));
            t = fmaxf(t, __shfl_xor(t, 4, 32));
            t = fmaxf(t, __shfl_xor(t, 8, 32));
            const float mn = fmaxf(mr[j], t);
            const float al = __expf(mr[j] - mn);
            const float p0 = __expf(s[0][j] - mn);
            const float p1 = __expf(s[1][j] - mn);
            s[0][j] = p0; s[1][j] = p1;
            float rs = p0 + p1;
            rs += __shfl_xor(rs, 1, 32);
            rs += __shfl_xor(rs, 2, 32);
            rs += __shfl_xor(rs, 4, 32);
            rs += __shfl_xor(rs, 8, 32);
            lr[j] = lr[j] * al + rs;
            mr[j] = mn;
            O[0][j] *= al; O[1][j] *= al; O[2][j] *= al; O[3][j] *= al;
        }
        // --- P: C-layout -> wave-private LDS (16x32) -> A-layout ---
#pragma unroll
        for (int j = 0; j < 8; ++j) {
            const int row = j + half * 8;
            ptile[wave][row][l16]      = (bf16)s[0][j];
            ptile[wave][row][16 + l16] = (bf16)s[1][j];
        }
        // intra-wave store->load fence (LDS is in-order per wave; this also
        // pins the compiler's ds op ordering via the memory clobber)
        asm volatile("s_wait_dscnt 0x0" ::: "memory");
        v8bf plo = *(const v8bf*)&ptile[wave][l16][half * 16];
        v8bf phi = *(const v8bf*)&ptile[wave][l16][half * 16 + 8];
        v16bf pa = cat8(plo, phi);
        // --- O += P * V : 4 n-tiles of hd ---
#pragma unroll
        for (int g = 0; g < 4; ++g) {
            const bf16* vr = vbase + (size_t)(g * 16 + l16) * TPAD + kt + half * 16;
            v16bf vb = *(const v16bf*)vr;
            O[g] = wmma_bf16(pa, vb, O[g]);
        }
    }

    // --- normalize and store to y[B,T,C] (head-concat), bf16 ---
#pragma unroll
    for (int j = 0; j < 8; ++j) {
        const int row = j + half * 8;
        const int t = m0 + row;
        if (t < TSEQ) {
            const float inv = 1.0f / lr[j];
            const size_t base = ((size_t)b * TSEQ + t) * CDIM + h * HDIM;
#pragma unroll
            for (int g = 0; g < 4; ++g)
                Y[base + g * 16 + l16] = (bf16)(O[g][j] * inv);
        }
    }
}

// ---------------------------------------------------------------------------
// Launcher
// ---------------------------------------------------------------------------
extern "C" void kernel_launch(void* const* d_in, const int* in_sizes, int n_in,
                              void* d_out, int out_size, void* d_ws, size_t ws_size,
                              hipStream_t stream) {
    const float* x     = (const float*)d_in[0];
    const float* Wq    = (const float*)d_in[1];
    const float* Wk    = (const float*)d_in[2];
    const float* Wv    = (const float*)d_in[3];
    const float* Wproj = (const float*)d_in[4];
    const float* qn_w  = (const float*)d_in[5];
    const float* qn_b  = (const float*)d_in[6];
    const float* kn_w  = (const float*)d_in[7];
    const float* kn_b  = (const float*)d_in[8];
    const float* fcos  = (const float*)d_in[9];
    const float* fsin  = (const float*)d_in[10];
    float* out = (float*)d_out;

    // workspace carve
    char* p = (char*)d_ws;
    auto carve = [&](size_t bytes) {
        void* r = (void*)p;
        p += (bytes + 255) & ~(size_t)255;
        return r;
    };
    const size_t xw_elems = (size_t)MROWS * CDIM;          // 12,693,504
    const size_t ww_elems = (size_t)CDIM * CDIM;           // 589,824
    const size_t hm_elems = (size_t)BATCH * NHEAD * TPAD * HDIM; // 13,369,344

    bf16*  x_bf   = (bf16*) carve(xw_elems * 2);
    bf16*  wq_bf  = (bf16*) carve(ww_elems * 2);
    bf16*  wk_bf  = (bf16*) carve(ww_elems * 2);
    bf16*  wv_bf  = (bf16*) carve(ww_elems * 2);
    bf16*  wp_bf  = (bf16*) carve(ww_elems * 2);
    float* q_raw  = (float*)carve(xw_elems * 4);
    float* k_raw  = (float*)carve(xw_elems * 4);
    float* v_raw  = (float*)carve(xw_elems * 4);
    bf16*  q_bf   = (bf16*) carve(hm_elems * 2);
    bf16*  k_bf   = (bf16*) carve(hm_elems * 2);
    bf16*  vT_bf  = (bf16*) carve(hm_elems * 2);
    bf16*  y_bf   = (bf16*)q_raw;   // q_raw is dead by the time attention writes y

    // 1) bf16 conversions
    {
        int n4 = (int)(xw_elems / 4);
        k_f32_to_bf16<<<(n4 + 255) / 256, 256, 0, stream>>>(x, x_bf, n4);
        int w4 = (int)(ww_elems / 4);
        k_f32_to_bf16<<<(w4 + 255) / 256, 256, 0, stream>>>(Wq,    wq_bf, w4);
        k_f32_to_bf16<<<(w4 + 255) / 256, 256, 0, stream>>>(Wk,    wk_bf, w4);
        k_f32_to_bf16<<<(w4 + 255) / 256, 256, 0, stream>>>(Wv,    wv_bf, w4);
        k_f32_to_bf16<<<(w4 + 255) / 256, 256, 0, stream>>>(Wproj, wp_bf, w4);
    }
    // 2) QKV projections (WMMA GEMM, async-LDS staged A)
    {
        dim3 grid(MROWS / 16, CDIM / 256);
        k_gemm_bf16nt<<<grid, 128, 0, stream>>>(x_bf, wq_bf, q_raw, MROWS, CDIM, CDIM);
        k_gemm_bf16nt<<<grid, 128, 0, stream>>>(x_bf, wk_bf, k_raw, MROWS, CDIM, CDIM);
        k_gemm_bf16nt<<<grid, 128, 0, stream>>>(x_bf, wv_bf, v_raw, MROWS, CDIM, CDIM);
    }
    // 3) zero-pad head-major buffers (pad rows must be 0 every call)
    {
        int n8 = (int)(hm_elems / 8);
        k_zero_bf16<<<(n8 + 255) / 256, 256, 0, stream>>>(q_bf,  n8);
        k_zero_bf16<<<(n8 + 255) / 256, 256, 0, stream>>>(k_bf,  n8);
        k_zero_bf16<<<(n8 + 255) / 256, 256, 0, stream>>>(vT_bf, n8);
    }
    // 4) LayerNorm + RoPE (q pre-scaled by 1/sqrt(hd) = 0.125)
    k_ln_rope<<<MROWS, 192, 0, stream>>>(q_raw, qn_w, qn_b, fcos, fsin, q_bf, 0.125f);
    k_ln_rope<<<MROWS, 192, 0, stream>>>(k_raw, kn_w, kn_b, fcos, fsin, k_bf, 1.0f);
    // 5) V transpose to [B,H,hd,TPAD]
    {
        long total = (long)BATCH * NHEAD * HDIM * TSEQ;
        int blocks = (int)((total + 255) / 256);
        k_transpose_v<<<blocks, 256, 0, stream>>>(v_raw, vT_bf, total);
    }
    // 6) attention
    {
        dim3 grid(TPAD / 64, BATCH * NHEAD);
        k_attention<<<grid, 128, 0, stream>>>(q_bf, k_bf, vT_bf, y_bf);
    }
    // 7) output projection straight into d_out (f32)
    {
        dim3 grid(MROWS / 16, CDIM / 256);
        k_gemm_bf16nt<<<grid, 128, 0, stream>>>(y_bf, wp_bf, out, MROWS, CDIM, CDIM);
    }
}
